// Conv3d_88364657148042
// MI455X (gfx1250) — matrix operand order repaired and checked by reading, not compiled, not run on hardware
//
#include <hip/hip_runtime.h>
#include <hip/hip_bf16.h>

// ---------------------------------------------------------------------------
// Sparse 3D conv (rulebook): out[j] = sum_k feats[nm[k,j]] @ W[k] + bias
// N=400000, Cin=Cout=64, KVOL=27.  bf16 WMMA (f32 accumulate).
// feats (102 MB) is L2-resident (192 MB L2) -> per-row gathers are L2 hits.
// Weights staged once per block into LDS via one TDM tensor_load_to_lds.
// Invalid rulebook entries gather from a zero page (no mask VALU needed).
// Each wave owns a 32(M) x 64(N) tile: every LDS B-fragment feeds 2 WMMAs.
// ---------------------------------------------------------------------------

#define KVOL 27
#define CIN 64
#define COUT 64
// Packed B layout: [k][chunk(2)][ntile(4)][lane(32)][vgpr(8)] u32 (bf16 pair)
#define PACKED_WORDS (KVOL * 2 * 4 * 32 * 8)      // 55296 u32 = 221184 bytes
#define ZROW_WORDS   64                           // 256B zero feature row
#define WS_WORDS     (PACKED_WORDS + ZROW_WORDS)
#define LDS_BYTES    (PACKED_WORDS * 4)

typedef __attribute__((ext_vector_type(16))) __bf16        v16bf;
typedef __attribute__((ext_vector_type(8)))  float         v8f;
typedef __attribute__((ext_vector_type(4))) unsigned int   u32x4;
typedef __attribute__((ext_vector_type(8)))  int           i32x8;
typedef __attribute__((ext_vector_type(4)))  int           i32x4;

// ---------------------------------------------------------------------------
// Kernel 1: pack f32 weights [27][64][64] -> bf16 WMMA-B fragment layout, and
// append a 256B zero row.  Fragment (k, chunk c, ntile t): lane L (col=L&15,
// half=L>>4) needs VGPR v = pack(W[k][K][co], W[k][K+1][co]) with
// K = c*32 + half*16 + 2v, co = t*16 + col, contiguous -> 2x ds_load_b128.
// ---------------------------------------------------------------------------
__global__ void pack_weights_kernel(const float* __restrict__ w,
                                    unsigned int* __restrict__ ws) {
  int s = blockIdx.x * blockDim.x + threadIdx.x;
  if (s >= WS_WORDS) return;
  if (s >= PACKED_WORDS) { ws[s] = 0u; return; }   // zero feature row
  int v    = s & 7;
  int L    = (s >> 3) & 31;
  int t    = (s >> 8) & 3;
  int c    = (s >> 10) & 1;
  int k    = s >> 11;
  int col  = L & 15;
  int half = L >> 4;
  int K    = c * 32 + half * 16 + 2 * v;
  int co   = t * 16 + col;
  float x0 = w[k * (CIN * COUT) + K * COUT + co];
  float x1 = w[k * (CIN * COUT) + (K + 1) * COUT + co];
  union { __bf16 h[2]; unsigned int u; } p;
  p.h[0] = (__bf16)x0;
  p.h[1] = (__bf16)x1;
  ws[s] = p.u;
}

// ---------------------------------------------------------------------------
// Kernel 2: main conv. 256 threads = 8 waves; wave -> 32 rows x 64 cols.
// ---------------------------------------------------------------------------
extern __shared__ unsigned int lds_b[];

__device__ __forceinline__ v16bf cvt16(float4 f0, float4 f1, float4 f2,
                                       float4 f3) {
  v16bf a;
  a[0]  = (__bf16)f0.x; a[1]  = (__bf16)f0.y;
  a[2]  = (__bf16)f0.z; a[3]  = (__bf16)f0.w;
  a[4]  = (__bf16)f1.x; a[5]  = (__bf16)f1.y;
  a[6]  = (__bf16)f1.z; a[7]  = (__bf16)f1.w;
  a[8]  = (__bf16)f2.x; a[9]  = (__bf16)f2.y;
  a[10] = (__bf16)f2.z; a[11] = (__bf16)f2.w;
  a[12] = (__bf16)f3.x; a[13] = (__bf16)f3.y;
  a[14] = (__bf16)f3.z; a[15] = (__bf16)f3.w;
  return a;
}

__global__ void __launch_bounds__(256)
conv3d_wmma_kernel(const float* __restrict__ feats,
                   const float* __restrict__ bias,
                   const long long* __restrict__ nm,
                   const unsigned int* __restrict__ ws,
                   float* __restrict__ out, int n) {
  const int lane  = threadIdx.x & 31;
  const int wave  = threadIdx.x >> 5;
  const int mrow  = lane & 15;   // M index within 16-row A tile
  const int khalf = lane >> 4;   // which K half this lane holds
  const float* zrow = (const float*)(ws + PACKED_WORDS);

  // ---- Stage all packed weights (216 KB) into LDS via TDM (wave 0 only) ----
#if defined(__has_builtin) && __has_builtin(__builtin_amdgcn_tensor_load_to_lds)
  if (threadIdx.x < 32) {
    unsigned long long ga = (unsigned long long)(const void*)ws;
    unsigned int lds_off  = (unsigned int)(unsigned long long)&lds_b[0];
    const unsigned int elems = PACKED_WORDS / 2;  // 27648 x 8-byte elements
    u32x4 g0;
    g0[0] = 1u;                                     // count=1, user descriptor
    g0[1] = lds_off;                                // lds_addr
    g0[2] = (unsigned int)ga;                       // global_addr[31:0]
    g0[3] = (unsigned int)(ga >> 32) | 0x80000000u; // addr[56:32] | type=2
    i32x8 g1;
    g1[0] = (int)(3u << 16);                        // data_size = 3 (8 bytes)
    g1[1] = (int)((elems & 0xFFFFu) << 16);         // tensor_dim0 lo16
    g1[2] = (int)((elems >> 16) | (1u << 16));      // tensor_dim0 hi16|dim1=1
    g1[3] = (int)(elems << 16);                     // dim1 hi16=0 | tile_dim0
    g1[4] = 1;                                      // tile_dim1 = 1
    g1[5] = (int)elems;                             // tensor_dim0_stride lo32
    g1[6] = 0;
    g1[7] = 0;
    i32x4 g2 = {0, 0, 0, 0};
    i32x4 g3 = {0, 0, 0, 0};
    i32x8 g4 = {0, 0, 0, 0, 0, 0, 0, 0};            // unused trailing group
    __builtin_amdgcn_tensor_load_to_lds(g0, g1, g2, g3, g4, 0);
    __builtin_amdgcn_s_wait_tensorcnt(0);
  }
#else
  for (int i = threadIdx.x; i < PACKED_WORDS; i += 256) lds_b[i] = ws[i];
#endif
  __syncthreads();

  const int rowbase = blockIdx.x * 256 + wave * 32;
  const int row0    = rowbase + mrow;        // A tile 0
  const int row1    = rowbase + 16 + mrow;   // A tile 1
  const int rowc0   = (row0 < n) ? row0 : (n - 1);
  const int rowc1   = (row1 < n) ? row1 : (n - 1);

  v8f aA0 = {}, aA1 = {}, aA2 = {}, aA3 = {};   // rows rowbase..+15
  v8f aB0 = {}, aB1 = {}, aB2 = {}, aB3 = {};   // rows rowbase+16..+31

  typedef union { u32x4 u[2]; v16bf b; } Bfrag;

  for (int k = 0; k < KVOL; ++k) {
    long long idx0 = nm[(long long)k * n + rowc0];
    long long idx1 = nm[(long long)k * n + rowc1];
    if (k + 1 < KVOL) {
      __builtin_prefetch(&nm[(long long)(k + 1) * n + rowc0], 0, 0);
      __builtin_prefetch(&nm[(long long)(k + 1) * n + rowc1], 0, 0);
    }
    const float* s0 =
        (idx0 >= 0) ? feats + (unsigned long long)idx0 * CIN : zrow;
    const float* s1 =
        (idx1 >= 0) ? feats + (unsigned long long)idx1 * CIN : zrow;

#pragma unroll
    for (int c = 0; c < 2; ++c) {
      // ---- A fragments: 16x32 bf16.  Lane holds its row,
      // elems 0..15 = K kb..kb+15 ----
      const int kb = c * 32 + khalf * 16;
      const float4* p0 = (const float4*)(s0 + kb);
      const float4* p1 = (const float4*)(s0 + kb + 8);
      const float4* q0 = (const float4*)(s1 + kb);
      const float4* q1 = (const float4*)(s1 + kb + 8);
      v16bf a0 = cvt16(p0[0], p0[1], p1[0], p1[1]);
      v16bf a1 = cvt16(q0[0], q0[1], q1[0], q1[1]);

      // ---- B fragments, double-buffered; each feeds two WMMAs ----
#define LOAD_B(FR, T)                                                         \
      {                                                                       \
        const u32x4* bq = (const u32x4*)&lds_b[(((k * 2 + c) * 4 + (T)) * 32  \
                                                + lane) * 8];                 \
        FR.u[0] = bq[0];                                                      \
        FR.u[1] = bq[1];                                                      \
      }
#define WMMA(ACC, A, B)                                                       \
      ACC = __builtin_amdgcn_wmma_f32_16x16x32_bf16(false, A, false, B,       \
                                                    (short)0, ACC, false,     \
                                                    false);
      Bfrag b0, b1;
      LOAD_B(b0, 0)
      LOAD_B(b1, 1)
      WMMA(aA0, a0, b0.b)
      WMMA(aB0, a1, b0.b)
      LOAD_B(b0, 2)
      WMMA(aA1, a0, b1.b)
      WMMA(aB1, a1, b1.b)
      LOAD_B(b1, 3)
      WMMA(aA2, a0, b0.b)
      WMMA(aB2, a1, b0.b)
      WMMA(aA3, a0, b1.b)
      WMMA(aB3, a1, b1.b)
#undef LOAD_B
#undef WMMA
    }
  }

  // ---- Epilogue: C layout (16x16 f32): VGPR r -> M = r (lanes 0-15),
  // M = 8+r (lanes 16-31); N = lane&15 ----
  const int col = mrow;
  const float c0 = bias[col];
  const float c1 = bias[16 + col];
  const float c2 = bias[32 + col];
  const float c3 = bias[48 + col];
  const int mbase = rowbase + (khalf ? 8 : 0);
#pragma unroll
  for (int r = 0; r < 8; ++r) {
    const int m0 = mbase + r;
    if (m0 < n) {
      float* o = out + (unsigned long long)m0 * COUT;
      o[col]      = aA0[r] + c0;
      o[16 + col] = aA1[r] + c1;
      o[32 + col] = aA2[r] + c2;
      o[48 + col] = aA3[r] + c3;
    }
    const int m1 = mbase + 16 + r;
    if (m1 < n) {
      float* o = out + (unsigned long long)m1 * COUT;
      o[col]      = aB0[r] + c0;
      o[16 + col] = aB1[r] + c1;
      o[32 + col] = aB2[r] + c2;
      o[48 + col] = aB3[r] + c3;
    }
  }
}

// ---------------------------------------------------------------------------
extern "C" void kernel_launch(void* const* d_in, const int* in_sizes, int n_in,
                              void* d_out, int out_size, void* d_ws, size_t ws_size,
                              hipStream_t stream) {
  const float*     feats = (const float*)d_in[0];
  const float*     kern  = (const float*)d_in[1];
  const float*     bias  = (const float*)d_in[2];
  const long long* nm    = (const long long*)d_in[3];
  float*           out   = (float*)d_out;
  unsigned int*    wsB   = (unsigned int*)d_ws;
  const int n = in_sizes[0] / CIN;

  (void)hipFuncSetAttribute((const void*)conv3d_wmma_kernel,
                            hipFuncAttributeMaxDynamicSharedMemorySize,
                            LDS_BYTES);

  pack_weights_kernel<<<(WS_WORDS + 255) / 256, 256, 0, stream>>>(kern, wsB);

  const int grid = (n + 255) / 256;
  conv3d_wmma_kernel<<<grid, 256, LDS_BYTES, stream>>>(feats, bias, nm, wsB,
                                                       out, n);
}
